// MultiGRU_71055938945257
// MI455X (gfx1250) — matrix operand order repaired and checked by reading, not compiled, not run on hardware
//
#include <hip/hip_runtime.h>
#include <hip/hip_bf16.h>

// ---------------------------------------------------------------------------
// MultiGRU on MI455X (gfx1250): bf16 WMMA GEMMs (register-blocked 32x64 per
// wave, preconverted bf16 operands) + persistent grid-sync recurrent scan.
// ---------------------------------------------------------------------------

#define TSEQ   2048
#define HID    1024
#define EMB    1024
#define G3H    3072
#define NLAY   10
#define NTAGS  50257

typedef __attribute__((ext_vector_type(16))) __bf16 v16bf;
typedef __attribute__((ext_vector_type(8)))  float  v8f;
typedef unsigned short ushort_t;

union BFV { v16bf v; uint4 q[2]; unsigned u[8]; };

__device__ __forceinline__ ushort_t f2bf(float f) {          // RNE fp32->bf16
    unsigned u = __builtin_bit_cast(unsigned, f);
    unsigned r = u + 0x7FFFu + ((u >> 16) & 1u);
    return (ushort_t)(r >> 16);
}
__device__ __forceinline__ float bflo(unsigned u) { return __builtin_bit_cast(float, u << 16); }
__device__ __forceinline__ float bfhi(unsigned u) { return __builtin_bit_cast(float, u & 0xFFFF0000u); }

// ---- WMMA tile loads from bf16 memory, contiguous-k layouts (k = 16*half + s) -----------------
// A 16x32 (MxK): lane<16 -> row=lane,    K = k..k+15
//                lane>=16 -> row=lane-16, K = k+16..k+31
__device__ __forceinline__ v16bf load_tileA_bf(const ushort_t* __restrict__ A, int lda,
                                               int mBase, int kk, int lane) {
    int r = lane & 15, half = lane >> 4;
    const ushort_t* p = A + (size_t)(mBase + r) * lda + kk + half * 16;
    BFV u;
    u.q[0] = *(const uint4*)(p);        // K chunk {k0..k0+7}
    u.q[1] = *(const uint4*)(p + 8);    // K chunk {k0+8..k0+15}
    return u.v;
}
// B 32x16 (KxN) = W^T, W row-major [N][K]:
// lane<16 -> col=lane, K = k..k+15 ; lane>=16 -> col=lane-16, K = k+16..k+31
__device__ __forceinline__ v16bf load_tileB_bf(const ushort_t* __restrict__ W, int ldb,
                                               int nBase, int kk, int lane, int Nmax) {
    int c = lane & 15, half = lane >> 4;
    int row = nBase + c; if (row >= Nmax) row = Nmax - 1;    // clamp tail tiles
    const ushort_t* p = W + (size_t)row * ldb + kk + half * 16;
    BFV u;
    u.q[0] = *(const uint4*)(p);
    u.q[1] = *(const uint4*)(p + 8);
    return u.v;
}
// Fallback: B from fp32 memory, packed to bf16 with one v_perm per 2 elems.
__device__ __forceinline__ v16bf load_tileB_f32(const float* __restrict__ W, int ldb,
                                                int nBase, int kk, int lane, int Nmax) {
    int c = lane & 15, half = lane >> 4;
    int row = nBase + c; if (row >= Nmax) row = Nmax - 1;
    const float* p = W + (size_t)row * ldb + kk + half * 16;
    BFV u;
#pragma unroll
    for (int i = 0; i < 4; ++i) {
        float4 f = *(const float4*)(p + i * 4);
        unsigned a0 = __builtin_bit_cast(unsigned, f.x);
        unsigned a1 = __builtin_bit_cast(unsigned, f.y);
        unsigned a2 = __builtin_bit_cast(unsigned, f.z);
        unsigned a3 = __builtin_bit_cast(unsigned, f.w);
        u.u[i * 2]     = __builtin_amdgcn_perm(a1, a0, 0x07060302u); // {hi16(a1),hi16(a0)}
        u.u[i * 2 + 1] = __builtin_amdgcn_perm(a3, a2, 0x07060302u);
    }
    return u.v;
}

__device__ __forceinline__ void store_ctile(float* __restrict__ C, int N,
                                            const v8f acc[2][4],
                                            int mBase, int nBase0, int lane,
                                            const float* __restrict__ bias) {
    int colOff = lane & 15, rowAdd = (lane >> 4) * 8;
#pragma unroll
    for (int j = 0; j < 4; ++j) {
        int col = nBase0 + j * 16 + colOff;
        if (col < N) {
            float bv = bias[col];
#pragma unroll
            for (int i = 0; i < 2; ++i)
#pragma unroll
                for (int r = 0; r < 8; ++r)
                    C[(size_t)(mBase + i * 16 + rowAdd + r) * N + col] = acc[i][j][r] + bv;
        }
    }
}

// ---- C = A[MxK]*B[NxK]^T + bias; A,B bf16; fp32 accumulate/out ------------
// block = 8 waves = 256 rows x 64 cols; wave = 32x64 (2x4 WMMA tiles)
__global__ __launch_bounds__(256)
void gemm_bf16(const ushort_t* __restrict__ A, const ushort_t* __restrict__ B,
               const float* __restrict__ bias, float* __restrict__ C,
               int M, int N, int K, int tilesN64) {
    int bm = blockIdx.x / tilesN64, bn = blockIdx.x % tilesN64;
    int wave = threadIdx.x >> 5, lane = threadIdx.x & 31;
    int mBase = bm * 256 + wave * 32, nBase0 = bn * 64;

    v8f acc[2][4];
#pragma unroll
    for (int i = 0; i < 2; ++i)
#pragma unroll
        for (int j = 0; j < 4; ++j) acc[i][j] = (v8f){0.f,0.f,0.f,0.f,0.f,0.f,0.f,0.f};

    for (int kk = 0; kk < K; kk += 32) {
        if (kk + 32 < K)   // next B strip -> global_prefetch_b8
            __builtin_prefetch(B + (size_t)(nBase0 + (lane & 15)) * K + kk + 32, 0, 1);
        // issue all tile loads into distinct regs, then the 8 WMMAs
        v16bf a0 = load_tileA_bf(A, K, mBase,      kk, lane);
        v16bf a1 = load_tileA_bf(A, K, mBase + 16, kk, lane);
        v16bf b[4];
#pragma unroll
        for (int j = 0; j < 4; ++j)
            b[j] = load_tileB_bf(B, K, nBase0 + j * 16, kk, lane, N);
#pragma unroll
        for (int j = 0; j < 4; ++j) {
            acc[0][j] = __builtin_amdgcn_wmma_f32_16x16x32_bf16(
                false, a0, false, b[j], (short)0, acc[0][j], false, false);
            acc[1][j] = __builtin_amdgcn_wmma_f32_16x16x32_bf16(
                false, a1, false, b[j], (short)0, acc[1][j], false, false);
        }
    }
    store_ctile(C, N, acc, mBase, nBase0, lane, bias);
}

// Same GEMM, B read from fp32 memory (used only when ws can't hold bf16 Wout)
__global__ __launch_bounds__(256)
void gemm_bf16_wf32(const ushort_t* __restrict__ A, const float* __restrict__ B,
                    const float* __restrict__ bias, float* __restrict__ C,
                    int M, int N, int K, int tilesN64) {
    int bm = blockIdx.x / tilesN64, bn = blockIdx.x % tilesN64;
    int wave = threadIdx.x >> 5, lane = threadIdx.x & 31;
    int mBase = bm * 256 + wave * 32, nBase0 = bn * 64;
    v8f acc[2][4];
#pragma unroll
    for (int i = 0; i < 2; ++i)
#pragma unroll
        for (int j = 0; j < 4; ++j) acc[i][j] = (v8f){0.f,0.f,0.f,0.f,0.f,0.f,0.f,0.f};
    for (int kk = 0; kk < K; kk += 32) {
        v16bf a0 = load_tileA_bf(A, K, mBase,      kk, lane);
        v16bf a1 = load_tileA_bf(A, K, mBase + 16, kk, lane);
        v16bf b[4];
#pragma unroll
        for (int j = 0; j < 4; ++j)
            b[j] = load_tileB_f32(B, K, nBase0 + j * 16, kk, lane, N);
#pragma unroll
        for (int j = 0; j < 4; ++j) {
            acc[0][j] = __builtin_amdgcn_wmma_f32_16x16x32_bf16(
                false, a0, false, b[j], (short)0, acc[0][j], false, false);
            acc[1][j] = __builtin_amdgcn_wmma_f32_16x16x32_bf16(
                false, a1, false, b[j], (short)0, acc[1][j], false, false);
        }
    }
    store_ctile(C, N, acc, mBase, nBase0, lane, bias);
}

// ---- grid-wide generation barrier -----------------------------------------
#define SCAN_BLOCKS 64
__device__ __forceinline__ void grid_barrier(unsigned* cnt, unsigned* gen, unsigned nblk) {
    __syncthreads();
    if (threadIdx.x == 0) {
        __threadfence();
        unsigned g = __hip_atomic_load(gen, __ATOMIC_RELAXED, __HIP_MEMORY_SCOPE_AGENT);
        unsigned a = __hip_atomic_fetch_add(cnt, 1u, __ATOMIC_ACQ_REL, __HIP_MEMORY_SCOPE_AGENT);
        if (a == nblk - 1u) {
            __hip_atomic_store(cnt, 0u, __ATOMIC_RELAXED, __HIP_MEMORY_SCOPE_AGENT);
            __hip_atomic_store(gen, g + 1u, __ATOMIC_RELEASE, __HIP_MEMORY_SCOPE_AGENT);
        } else {
            while (__hip_atomic_load(gen, __ATOMIC_ACQUIRE, __HIP_MEMORY_SCOPE_AGENT) == g)
                __builtin_amdgcn_s_sleep(2);
        }
    }
    __syncthreads();
}

// dot(w_bf16[1024], h_fp32_lds[1024]) -> full sum in every lane
__device__ __forceinline__ float rowdot_bf(const ushort_t* __restrict__ w,
                                           const float* hs, int lane) {
    float acc = 0.f;
#pragma unroll
    for (int i = 0; i < 4; ++i) {
        int e = i * 256 + lane * 8;
        uint4  wq = *(const uint4*)(w + e);
        float4 h0 = *(const float4*)(hs + e);
        float4 h1 = *(const float4*)(hs + e + 4);
        acc += bflo(wq.x) * h0.x + bfhi(wq.x) * h0.y
             + bflo(wq.y) * h0.z + bfhi(wq.y) * h0.w
             + bflo(wq.z) * h1.x + bfhi(wq.z) * h1.y
             + bflo(wq.w) * h1.z + bfhi(wq.w) * h1.w;
    }
#pragma unroll
    for (int m = 16; m >= 1; m >>= 1) acc += __shfl_xor(acc, m, 32);
    return acc;
}

// ---- persistent GRU scan: 64 blocks x 8 waves; wave owns 2 hidden units ---
__global__ __launch_bounds__(256)
void gru_scan(const float* __restrict__ gx, const ushort_t* __restrict__ Whh,
              const float* __restrict__ bhh, const float* __restrict__ h0l,
              ushort_t* __restrict__ xout, float* __restrict__ hbuf,
              unsigned* cnt, unsigned* gen) {
    __shared__ __align__(16) float hs[HID];
    int lane = threadIdx.x & 31;
    int wave = threadIdx.x >> 5;
    int gw   = blockIdx.x * 8 + wave;   // 0..511
    int j0   = gw * 2;

    if (lane < 2) hbuf[j0 + lane] = h0l[j0 + lane];
    grid_barrier(cnt, gen, SCAN_BLOCKS);

    for (int t = 0; t < TSEQ; ++t) {
        for (int i = threadIdx.x; i < HID; i += 256) hs[i] = hbuf[i];
        __syncthreads();

        const float* gxt = gx + (size_t)t * G3H;
        float hnew[2];
#pragma unroll
        for (int u = 0; u < 2; ++u) {
            int j = j0 + u;
            float ghr = rowdot_bf(Whh + (size_t)j           * HID, hs, lane) + bhh[j];
            float ghz = rowdot_bf(Whh + (size_t)(j + HID)   * HID, hs, lane) + bhh[j + HID];
            float ghn = rowdot_bf(Whh + (size_t)(j + 2*HID) * HID, hs, lane) + bhh[j + 2*HID];
            float r = 1.f / (1.f + expf(-(gxt[j]       + ghr)));
            float z = 1.f / (1.f + expf(-(gxt[j + HID] + ghz)));
            float n = tanhf(gxt[j + 2*HID] + r * ghn);
            hnew[u] = (1.f - z) * n + z * hs[j];
        }
        if (lane == 0) {
            hbuf[j0]     = hnew[0];
            hbuf[j0 + 1] = hnew[1];
            xout[(size_t)t * HID + j0]     = f2bf(hnew[0]);
            xout[(size_t)t * HID + j0 + 1] = f2bf(hnew[1]);
        }
        grid_barrier(cnt, gen, SCAN_BLOCKS);
    }
}

// ---- helpers --------------------------------------------------------------
__global__ void init_bar(unsigned* bar) { if (threadIdx.x < 2) bar[threadIdx.x] = 0u; }

__global__ __launch_bounds__(256)
void cvt_f32_bf16(const float* __restrict__ in, ushort_t* __restrict__ out, long n) {
    long i = ((long)blockIdx.x * 256 + threadIdx.x) * 4;
    if (i + 4 <= n) {
        float4 f = *(const float4*)(in + i);
        uint2 p;
        p.x = ((unsigned)f2bf(f.y) << 16) | f2bf(f.x);
        p.y = ((unsigned)f2bf(f.w) << 16) | f2bf(f.z);
        *(uint2*)(out + i) = p;
    }
}

__global__ void embed_kernel(const int* __restrict__ sent, const float* __restrict__ emb,
                             ushort_t* __restrict__ x) {
    int idx = blockIdx.x * blockDim.x + threadIdx.x;
    int t = idx >> 10, e = idx & (EMB - 1);
    x[idx] = f2bf(emb[(size_t)sent[t] * EMB + e]);
}

__global__ __launch_bounds__(256)
void lse_kernel(const float* __restrict__ logits, float* __restrict__ lse, int N) {
    __shared__ float red[256];
    const float* row = logits + (size_t)blockIdx.x * N;
    float mx = -INFINITY;
    for (int i = threadIdx.x; i < N; i += 256) mx = fmaxf(mx, row[i]);
    red[threadIdx.x] = mx; __syncthreads();
    for (int s = 128; s; s >>= 1) {
        if (threadIdx.x < s) red[threadIdx.x] = fmaxf(red[threadIdx.x], red[threadIdx.x + s]);
        __syncthreads();
    }
    mx = red[0]; __syncthreads();
    float sum = 0.f;
    for (int i = threadIdx.x; i < N; i += 256) sum += expf(row[i] - mx);
    red[threadIdx.x] = sum; __syncthreads();
    for (int s = 128; s; s >>= 1) {
        if (threadIdx.x < s) red[threadIdx.x] += red[threadIdx.x + s];
        __syncthreads();
    }
    if (threadIdx.x == 0) lse[blockIdx.x] = mx + logf(red[0]);
}

__global__ void sub_lse(float* __restrict__ out, const float* __restrict__ lse, int N) {
    int col = blockIdx.x * blockDim.x + threadIdx.x;
    if (col < N) out[(size_t)blockIdx.y * N + col] -= lse[blockIdx.y];
}

// ---------------------------------------------------------------------------
extern "C" void kernel_launch(void* const* d_in, const int* in_sizes, int n_in,
                              void* d_out, int out_size, void* d_ws, size_t ws_size,
                              hipStream_t stream) {
    const int*   sent = (const int*)  d_in[0];
    const float* h0   = (const float*)d_in[1];
    const float* emb  = (const float*)d_in[2];
    const float* Wih  = (const float*)d_in[3];
    const float* Whh  = (const float*)d_in[4];
    const float* bih  = (const float*)d_in[5];
    const float* bhh  = (const float*)d_in[6];
    const float* Wout = (const float*)d_in[7];
    const float* bout = (const float*)d_in[8];
    float* out = (float*)d_out;

    // ws layout: gx f32 | lse f32 | hbuf f32 | bar | xa/xb bf16 | Wih/Whh bf16 | [Wout bf16]
    char* base = (char*)d_ws;
    float*    gx     = (float*)base;                       // 2048*3072*4
    float*    lse    = gx + (size_t)TSEQ * G3H;
    float*    hbuf   = lse + TSEQ;
    unsigned* bar    = (unsigned*)(hbuf + HID);
    ushort_t* xa     = (ushort_t*)((char*)bar + 256);
    ushort_t* xb     = xa + (size_t)TSEQ * HID;
    ushort_t* wih_bf = xb + (size_t)TSEQ * HID;
    ushort_t* whh_bf = wih_bf + (size_t)G3H * EMB;
    ushort_t* wout_bf = whh_bf + (size_t)G3H * HID;
    size_t need_wout = (size_t)((char*)(wout_bf + (size_t)NTAGS * HID) - base);
    bool haveWoutBuf = ws_size >= need_wout;               // fixed per run -> deterministic

    init_bar<<<1, 32, 0, stream>>>(bar);
    embed_kernel<<<(TSEQ * EMB) / 256, 256, 0, stream>>>(sent, emb, xa);

    ushort_t* xc = xa;
    ushort_t* xn = xb;
    const long wsz = (long)G3H * EMB;
    for (int l = 0; l < NLAY; ++l) {
        cvt_f32_bf16<<<(unsigned)((wsz / 4 + 255) / 256), 256, 0, stream>>>(
            Wih + (size_t)l * wsz, wih_bf, wsz);
        cvt_f32_bf16<<<(unsigned)((wsz / 4 + 255) / 256), 256, 0, stream>>>(
            Whh + (size_t)l * wsz, whh_bf, wsz);
        // gx = xc @ Wih_l^T + bih_l   [2048 x 3072], bf16 WMMA
        gemm_bf16<<<(TSEQ / 256) * (G3H / 64), 256, 0, stream>>>(
            xc, wih_bf, bih + (size_t)l * G3H, gx, TSEQ, G3H, EMB, G3H / 64);
        // sequential scan: whole chip per GEMV, grid barrier per timestep
        gru_scan<<<SCAN_BLOCKS, 256, 0, stream>>>(
            gx, whh_bf, bhh + (size_t)l * G3H, h0 + (size_t)l * HID,
            xn, hbuf, bar, bar + 1);
        ushort_t* tmp = xc; xc = xn; xn = tmp;
    }

    // logits = xc @ Wout^T + bout -> d_out, then in-place log_softmax
    int tilesN64 = (NTAGS + 63) / 64;                      // 786
    if (haveWoutBuf) {
        const long on = (long)NTAGS * HID;
        cvt_f32_bf16<<<(unsigned)((on / 4 + 255) / 256), 256, 0, stream>>>(Wout, wout_bf, on);
        gemm_bf16<<<(TSEQ / 256) * tilesN64, 256, 0, stream>>>(
            xc, wout_bf, bout, out, TSEQ, NTAGS, HID, tilesN64);
    } else {
        gemm_bf16_wf32<<<(TSEQ / 256) * tilesN64, 256, 0, stream>>>(
            xc, Wout, bout, out, TSEQ, NTAGS, HID, tilesN64);
    }
    lse_kernel<<<TSEQ, 256, 0, stream>>>(out, lse, NTAGS);
    dim3 sg((NTAGS + 255) / 256, TSEQ);
    sub_lse<<<sg, 256, 0, stream>>>(out, lse, NTAGS);
}
